// DownConvLayers_30683246363152
// MI455X (gfx1250) — matrix operand order repaired and checked by reading, not compiled, not run on hardware
//
#include <hip/hip_runtime.h>
#include <hip/hip_bf16.h>

typedef __bf16 bf16_t;
typedef __attribute__((ext_vector_type(16))) __bf16 v16bf;
typedef __attribute__((ext_vector_type(8)))  __bf16 v8bf;
typedef __attribute__((ext_vector_type(4)))  __bf16 v4bf;
typedef __attribute__((ext_vector_type(8)))  float  v8f;

// fp32 -> bf16 round-to-nearest-even (bit trick)
static __device__ inline bf16_t f2bf(float f) {
    union { float f; unsigned u; } v; v.f = f;
    unsigned r = v.u + 0x7FFFu + ((v.u >> 16) & 1u);
    unsigned short h = (unsigned short)(r >> 16);
    bf16_t b;
    __builtin_memcpy(&b, &h, sizeof(b));
    return b;
}

// ---------------- degree / dinv (once; edges shared by all layers) ----------
__global__ void k_zero_f32(float* __restrict__ p, int n) {
    int i = blockIdx.x * blockDim.x + threadIdx.x;
    if (i < n) p[i] = 0.0f;
}

__global__ void k_degree(const int* __restrict__ dst, float* __restrict__ deg, int E) {
    int e = blockIdx.x * blockDim.x + threadIdx.x;
    if (e < E) atomicAdd(&deg[dst[e]], 1.0f);
}

__global__ void k_dinv(float* __restrict__ d, int N) {
    int i = blockIdx.x * blockDim.x + threadIdx.x;
    if (i < N) d[i] = rsqrtf(d[i] + 1.0f);   // +1 = self-loop
}

// ---------------- fp32 -> bf16 staging (4 elems/thread, 8B stores) ----------
__global__ void k_cvt_bf16(const float* __restrict__ in, bf16_t* __restrict__ out, long n4) {
    long i = (long)blockIdx.x * blockDim.x + threadIdx.x;
    if (i < n4) {
        float4 v = ((const float4*)in)[i];
        v4bf o;
        o[0] = f2bf(v.x); o[1] = f2bf(v.y); o[2] = f2bf(v.z); o[3] = f2bf(v.w);
        ((v4bf*)out)[i] = o;
    }
}

// ---------------- pack W[K,Nout] fp32 into bf16 B-fragment order ------------
// dst index = ((s*tilesN + tn)*32 + lane)*16 + (2h+j)
// lane layout (16-bit B 32x16): K = g*8 + q (q<8), 16 + g*8 + (q-8) (q>=8), N = lane&15
__global__ void k_packB(const float* __restrict__ Wm, bf16_t* __restrict__ pB,
                        int K, int Nout) {
    int idx = blockIdx.x * blockDim.x + threadIdx.x;
    if (idx >= K * Nout) return;
    int t = idx;
    int q    = t & 15;  t >>= 4;     // q = 2h+j
    int lane = t & 31;  t >>= 5;
    int tilesN = Nout >> 4;
    int tn = t % tilesN;
    int s  = t / tilesN;
    int g = lane >> 4, l = lane & 15;
    int k = s * 32 + g * 8 + q + ((q >> 3) << 3); // table order, matches A
    pB[idx] = f2bf(Wm[(long)k * Nout + tn * 16 + l]);
}

// ---------------- GEMM: Y[M,Nout] = Xb[M,K](bf16) @ packedB, f32 acc --------
// One wave per 16x16 tile; K-step 32. Inner loop: 4x b128 load + 1 WMMA.
__global__ void __launch_bounds__(256) k_gemm_wmma(const bf16_t* __restrict__ Xb,
                                                   const bf16_t* __restrict__ pB,
                                                   float* __restrict__ Y,
                                                   int M, int K, int Nout) {
    const int lane   = threadIdx.x & 31;
    const int wave   = threadIdx.x >> 5;
    const int tilesN = Nout >> 4;
    const int t      = blockIdx.x * 8 + wave;
    if (t >= (M >> 4) * tilesN) return;        // uniform per wave (EXEC stays full)

    const int tm = t / tilesN;
    const int tn = t % tilesN;
    const int g  = lane >> 4;
    const int l  = lane & 15;

    // A fragment: row = tm*16 + (lane&15); K spans [g*8, g*8+8) and +16
    const bf16_t* aRow = Xb + (long)(tm * 16 + l) * K + g * 8;
    // B fragment: 16 contiguous bf16 per lane in packed buffer
    const bf16_t* bPtr = pB + ((long)tn * 32 + lane) * 16;
    const long bStep = (long)tilesN * 512;     // per K-step advance

    v8f c = {0.f, 0.f, 0.f, 0.f, 0.f, 0.f, 0.f, 0.f};

    for (int k0 = 0; k0 < K; k0 += 32) {
        v8bf alo = *(const v8bf*)(aRow + k0);
        v8bf ahi = *(const v8bf*)(aRow + k0 + 16);
        v16bf a = __builtin_shufflevector(alo, ahi,
                    0,1,2,3,4,5,6,7,8,9,10,11,12,13,14,15);
        v8bf blo = *(const v8bf*)(bPtr);
        v8bf bhi = *(const v8bf*)(bPtr + 8);
        v16bf b = __builtin_shufflevector(blo, bhi,
                    0,1,2,3,4,5,6,7,8,9,10,11,12,13,14,15);
        bPtr += bStep;
        c = __builtin_amdgcn_wmma_f32_16x16x32_bf16(false, a, false, b,
                                                    (short)0, c, false, false);
    }

    // C/D: VGPR r -> M = r + g*8, N = lane&15.  M%16==0 -> no guards.
    float* yp = Y + (long)(tm * 16 + g * 8) * Nout + tn * 16 + l;
#pragma unroll
    for (int r = 0; r < 8; ++r) { *yp = c[r]; yp += Nout; }
}

// ---------------- self-loop init: agg = xw * dinv^2 + bias ------------------
__global__ void k_selfinit(const float* __restrict__ xw, const float* __restrict__ dinv,
                           const float* __restrict__ bias, float* __restrict__ agg,
                           long total, int lshF, int maskF) {
    long i = (long)blockIdx.x * blockDim.x + threadIdx.x;
    if (i < total) {
        int node = (int)(i >> lshF), f = (int)(i & maskF);
        float di = dinv[node];
        agg[i] = xw[i] * di * di + bias[f];
    }
}

// ---------------- edge scatter-add: agg[dst] += xw[src]*dinv[s]*dinv[d] -----
__global__ void k_scatter(const int* __restrict__ src, const int* __restrict__ dst,
                          const float* __restrict__ dinv, const float* __restrict__ xw,
                          float* __restrict__ agg, long total, int lshC, int maskC, int F) {
    long id = (long)blockIdx.x * blockDim.x + threadIdx.x;
    if (id >= total) return;
    int e  = (int)(id >> lshC);
    int cc = (int)(id & maskC);
    int s = src[e], d = dst[e];
    float coef = dinv[s] * dinv[d];
    float4 v = ((const float4*)(xw + (long)s * F))[cc];
    float* out = agg + (long)d * F + cc * 4;
    atomicAdd(out + 0, v.x * coef);
    atomicAdd(out + 1, v.y * coef);
    atomicAdd(out + 2, v.z * coef);
    atomicAdd(out + 3, v.w * coef);
}

__global__ void k_relu(float* __restrict__ p, long n) {
    long i = (long)blockIdx.x * blockDim.x + threadIdx.x;
    if (i < n) p[i] = fmaxf(p[i], 0.0f);
}

// ---------------------------------------------------------------------------
extern "C" void kernel_launch(void* const* d_in, const int* in_sizes, int n_in,
                              void* d_out, int out_size, void* d_ws, size_t ws_size,
                              hipStream_t stream) {
    const float* x    = (const float*)d_in[0];
    const int*   edge = (const int*)  d_in[1];   // [2,E]: row0=src, row1=dst
    const float* W[3] = { (const float*)d_in[2], (const float*)d_in[4], (const float*)d_in[6] };
    const float* B[3] = { (const float*)d_in[3], (const float*)d_in[5], (const float*)d_in[7] };

    const int H1 = in_sizes[3];              // 128
    const int C  = in_sizes[2] / H1;         // 256
    const int N  = in_sizes[0] / C;          // 100000
    const int E  = in_sizes[1] / 2;          // 800000
    const int H2 = in_sizes[5];              // 64
    const int H3 = in_sizes[7];              // 32

    const int* srcI = edge;
    const int* dstI = edge + E;

    // ws layout: dinv[N] f32 | bufA[N*H1] f32 | bufB[N*H1] f32 |
    //            bufX[N*C] bf16 | pB[C*H1] bf16
    float*  dinv = (float*)d_ws;
    float*  bufA = dinv + N;
    float*  bufB = bufA + (long)N * H1;
    bf16_t* bufX = (bf16_t*)(bufB + (long)N * H1);
    bf16_t* pB   = bufX + (long)N * C;
    float*  out  = (float*)d_out;

    const int T = 256;
    auto cdiv = [](long a, long b) { return (int)((a + b - 1) / b); };

    k_zero_f32<<<cdiv(N, T), T, 0, stream>>>(dinv, N);
    k_degree  <<<cdiv(E, T), T, 0, stream>>>(dstI, dinv, E);
    k_dinv    <<<cdiv(N, T), T, 0, stream>>>(dinv, N);

    const float* layerIn[3] = { x, bufB, bufB };
    float*       layerH[3]  = { bufB, bufB, out };
    const int    Kd[3] = { C,  H1, H2 };
    const int    Fd[3] = { H1, H2, H3 };

    for (int li = 0; li < 3; ++li) {
        const int Kl = Kd[li], Fl = Fd[li];
        const int lshF = __builtin_ctz(Fl);
        const int lshC = __builtin_ctz(Fl >> 2);

        // 1) stage input to bf16 (each element converted exactly once)
        long n4 = (long)N * Kl / 4;
        k_cvt_bf16<<<cdiv(n4, T), T, 0, stream>>>(layerIn[li], bufX, n4);
        // 2) pack weights into B-fragment order (tiny: <= 64 KB)
        k_packB<<<cdiv(Kl * Fl, T), T, 0, stream>>>(W[li], pB, Kl, Fl);
        // 3) WMMA GEMM -> bufA
        int tiles = (N / 16) * (Fl / 16);
        k_gemm_wmma<<<cdiv(tiles, 8), 256, 0, stream>>>(bufX, pB, bufA, N, Kl, Fl);
        // 4) self-loop + bias, 5) edge scatter, 6) ReLU
        long tot = (long)N * Fl;
        k_selfinit<<<cdiv(tot, T), T, 0, stream>>>(bufA, dinv, B[li], layerH[li],
                                                   tot, lshF, Fl - 1);
        long sc = (long)E * (Fl >> 2);
        k_scatter<<<cdiv(sc, T), T, 0, stream>>>(srcI, dstI, dinv, bufA, layerH[li],
                                                 sc, lshC, (Fl >> 2) - 1, Fl);
        k_relu<<<cdiv(tot, T), T, 0, stream>>>(layerH[li], tot);
    }
}
